// MHSA_67439576482275
// MI455X (gfx1250) — 16-bit and 8-bit matrix operand order repaired and checked by reading, compiled, not run on hardware
//
#include <hip/hip_runtime.h>
#include <stdint.h>

#define BN 2
#define TN 2048
#define DM 512
#define NH 8
#define DH 64

typedef __attribute__((ext_vector_type(16))) _Float16 v16h;
typedef __attribute__((ext_vector_type(8)))  _Float16 v8h;
typedef __attribute__((ext_vector_type(8)))  float    v8f;
typedef __attribute__((ext_vector_type(8)))  int      v8i;

#if defined(__gfx1250__) &&                                                    \
    __has_builtin(__builtin_amdgcn_global_load_async_to_lds_b128) &&           \
    __has_builtin(__builtin_amdgcn_s_wait_asynccnt)
#define USE_ASYNC_LDS 1
typedef int v4i32 __attribute__((vector_size(16)));
typedef __attribute__((address_space(1))) v4i32* gbl_v4p;
typedef __attribute__((address_space(3))) v4i32* lds_v4p;
#else
#define USE_ASYNC_LDS 0
#endif


static __device__ inline v16h load_a16(const _Float16* row, int g) {
  v8h lo = *(const v8h*)(row + g * 8);
  v8h hi = *(const v8h*)(row + 16 + g * 8);
  v16h r;
#pragma unroll
  for (int i = 0; i < 8; ++i) { r[i] = lo[i]; r[8 + i] = hi[i]; }
  return r;
}

static __device__ inline v16h load_b16(const _Float16* col, int g) {
  v8h lo = *(const v8h*)(col + g * 8);
  v8h hi = *(const v8h*)(col + 16 + g * 8);
  v16h r;
#pragma unroll
  for (int i = 0; i < 8; ++i) { r[i] = lo[i]; r[8 + i] = hi[i]; }
  return r;
}

static __device__ inline v8i load_a8(const int8_t* row, int g) {
  v8i a;
#pragma unroll
  for (int c = 0; c < 4; ++c) {
    const int* p = (const int*)(row + 16 * c + 8 * g);
    a[2 * c] = p[0]; a[2 * c + 1] = p[1];
  }
  return a;
}

static __device__ inline v8i load_b8(const int8_t* col, int g) {
  v8i b;
#pragma unroll
  for (int c = 0; c < 4; ++c) {
    const int* p = (const int*)(col + 16 * c + 8 * g);
    b[2 * c] = p[0]; b[2 * c + 1] = p[1];
  }
  return b;
}

static __device__ inline v8f wmma_f16(v16h a, v16h b, v8f c) {
  return __builtin_amdgcn_wmma_f32_16x16x32_f16(false, a, false, b, (short)0, c,
                                                false, false);
}

static __device__ inline float grp_max(float v) {
#pragma unroll
  for (int m = 8; m >= 1; m >>= 1) v = fmaxf(v, __shfl_xor(v, m, 32));
  return v;
}
static __device__ inline float grp_sum(float v) {
#pragma unroll
  for (int m = 8; m >= 1; m >>= 1) v += __shfl_xor(v, m, 32);
  return v;
}


__global__ __launch_bounds__(256) void k_wscale(const float* W0, const float* W1,
                                                const float* W2, const float* W3,
                                                const float* W4, float* wsbuf) {
  const float* Ws[5] = {W0, W1, W2, W3, W4};
  const float* W = Ws[blockIdx.x];
  __shared__ float red[256];
  float s = 0.f;
  for (int i = threadIdx.x; i < DM * DM; i += 256) s += fabsf(W[i]);
  red[threadIdx.x] = s;
  __syncthreads();
  for (int st = 128; st > 0; st >>= 1) {
    if ((int)threadIdx.x < st) red[threadIdx.x] += red[threadIdx.x + st];
    __syncthreads();
  }
  if (threadIdx.x == 0) wsbuf[blockIdx.x] = red[0] / (float)(DM * DM) + 1e-5f;
}

__global__ __launch_bounds__(256) void k_wquant(const float* W0, const float* W1,
                                                const float* W2, const float* W3,
                                                const float* W4,
                                                const float* __restrict__ wsbuf,
                                                int8_t* __restrict__ W8) {
  int i = blockIdx.x * 256 + threadIdx.x;
  int w = i >> 18;
  int off = i & 262143;
  const float* Ws[5] = {W0, W1, W2, W3, W4};
  float q = rintf(Ws[w][off] / wsbuf[w]);
  q = fminf(fmaxf(q, -1.f), 1.f);
  W8[i] = (int8_t)q;
}


__global__ __launch_bounds__(256) void k_lnq(const float* __restrict__ x,
                                             const float* __restrict__ g,
                                             const float* __restrict__ b,
                                             const int* __restrict__ bwp,
                                             int8_t* __restrict__ xq,
                                             float* __restrict__ dq, int mode) {
  int t = blockIdx.x;
  const float* row = x + (size_t)t * DM;
  __shared__ float r1[256];
  __shared__ float r2[256];
  int tid = threadIdx.x;
  float v0 = row[tid], v1 = row[tid + 256];
  float y0 = v0, y1 = v1;
  if (mode == 0) {
    r1[tid] = v0 + v1;
    r2[tid] = v0 * v0 + v1 * v1;
    __syncthreads();
    for (int st = 128; st > 0; st >>= 1) {
      if (tid < st) { r1[tid] += r1[tid + st]; r2[tid] += r2[tid + st]; }
      __syncthreads();
    }
    float mu = r1[0] * (1.f / (float)DM);
    float var = r2[0] * (1.f / (float)DM) - mu * mu;
    float rs = rsqrtf(var + 1e-5f);
    y0 = (v0 - mu) * rs * g[tid] + b[tid];
    y1 = (v1 - mu) * rs * g[tid + 256] + b[tid + 256];
    __syncthreads();
  }
  r1[tid] = fmaxf(fabsf(y0), fabsf(y1));
  __syncthreads();
  for (int st = 128; st > 0; st >>= 1) {
    if (tid < st) r1[tid] = fmaxf(r1[tid], r1[tid + st]);
    __syncthreads();
  }
  float Qb = exp2f((float)(bwp[0] - 1)) - 1.f;
  float xs = Qb / (r1[0] + 1e-5f);
  float q0 = fminf(fmaxf(rintf(y0 * xs), -Qb), Qb);
  float q1 = fminf(fmaxf(rintf(y1 * xs), -Qb), Qb);
  xq[(size_t)t * DM + tid] = (int8_t)q0;
  xq[(size_t)t * DM + tid + 256] = (int8_t)q1;
  if (tid == 0) dq[t] = 1.f / xs;
}


__global__ __launch_bounds__(128) void k_qgemm(const int8_t* __restrict__ A,
                                               const int8_t* __restrict__ W8,
                                               const float* __restrict__ dqA,
                                               const float* __restrict__ wsbuf,
                                               int widx,
                                               const float* __restrict__ bias,
                                               const float* __restrict__ resid,
                                               const int* __restrict__ maskp,
                                               float* __restrict__ out, int M) {
  __shared__ int8_t ldsA[16 * DM];
  int wid = threadIdx.x >> 5;
  int lane = threadIdx.x & 31;
  int g = lane >> 4, n = lane & 15;
  int tile = blockIdx.x * 4 + wid;
  int tm = tile >> 5;
  int tn = tile & 31;
  (void)M;

  {
    const int8_t* gsrc = A + (size_t)tm * 16 * DM;
    int tid = threadIdx.x;
#if USE_ASYNC_LDS
#pragma unroll
    for (int pass = 0; pass < 4; ++pass) {
      int off = pass * 2048 + tid * 16;
      __builtin_amdgcn_global_load_async_to_lds_b128(
          (gbl_v4p)(gsrc + off), (lds_v4p)(ldsA + off), 0, 0);
    }
    __builtin_amdgcn_s_wait_asynccnt(0);
#else
#pragma unroll
    for (int pass = 0; pass < 4; ++pass) {
      int off = pass * 2048 + tid * 16;
      *(int4*)(ldsA + off) = *(const int4*)(gsrc + off);
    }
#endif
    __syncthreads();
  }

  float ws = wsbuf[widx];
  const int8_t* arow = ldsA + (size_t)n * DM;
  const int8_t* bcol = W8 + (size_t)(tn * 16 + n) * DM;
  v8i acc = {0, 0, 0, 0, 0, 0, 0, 0};
#pragma unroll
  for (int k0 = 0; k0 < DM; k0 += 64) {
    __builtin_prefetch(bcol + k0 + 64, 0, 3);
    v8i a = load_a8(arow + k0, g);
    v8i bb = load_b8(bcol + k0, g);
    acc = __builtin_amdgcn_wmma_i32_16x16x64_iu8(true, a, true, bb, acc, false, false);
  }
  int col = tn * 16 + n;
  float bv = bias[col];
#pragma unroll
  for (int v = 0; v < 8; ++v) {
    int row = tm * 16 + v + 8 * g;
    float val = (float)acc[v] * ws * dqA[row] + bv;
    if (resid) {
      int bb2 = row >> 11, tt = row & (TN - 1);
      float mrow = (float)maskp[(size_t)bb2 * TN * TN + (size_t)tt * TN];
      val = resid[(size_t)row * DM + col] + val * mrow;
    }
    out[(size_t)row * DM + col] = val;
  }
}


__global__ __launch_bounds__(256) void k_pack(const float* __restrict__ Qf,
                                              const float* __restrict__ Kf,
                                              const float* __restrict__ Vf,
                                              const float* __restrict__ Pf,
                                              const float* __restrict__ pbu,
                                              const float* __restrict__ pbv,
                                              _Float16* __restrict__ Quh,
                                              _Float16* __restrict__ Qvh,
                                              _Float16* __restrict__ Kh,
                                              _Float16* __restrict__ Vt,
                                              _Float16* __restrict__ Ph) {
  int i = blockIdx.x * 256 + threadIdx.x;
  int c = i & (DM - 1);
  int tok = i >> 9;
  int b = tok >> 11, t = tok & (TN - 1);
  int h = c >> 6, d = c & (DH - 1);
  size_t bh = (size_t)(b * NH + h);
  size_t idx = (bh * TN + t) * DH + d;
  float q = Qf[i];
  Quh[idx] = (_Float16)(q + pbu[c]);
  Qvh[idx] = (_Float16)(q + pbv[c]);
  Kh[idx] = (_Float16)Kf[i];
  Vt[(bh * DH + d) * TN + t] = (_Float16)Vf[i];
  if (b == 0) Ph[((size_t)h * TN + t) * DH + d] = (_Float16)Pf[(size_t)t * DM + c];
}


__global__ __launch_bounds__(128) void k_attn(const _Float16* __restrict__ Quh,
                                              const _Float16* __restrict__ Qvh,
                                              const _Float16* __restrict__ Kh,
                                              const _Float16* __restrict__ Vt,
                                              const _Float16* __restrict__ Ph,
                                              const int* __restrict__ maskp,
                                              float* __restrict__ Hbuf) {
  __shared__ float bds[4][16][48];
  __shared__ _Float16 ps[4][16][32];
  int wid = threadIdx.x >> 5, lane = threadIdx.x & 31;
  int g = lane >> 4, ln = lane & 15;
  int wg = blockIdx.x * 4 + wid;
  int qt = wg & 127;
  int bh = wg >> 7;
  int b = bh >> 3, h = bh & 7;
  int q0 = qt * 16;

  const _Float16* Qub = Quh + ((size_t)bh * TN + q0 + ln) * DH;
  const _Float16* Qvb = Qvh + ((size_t)bh * TN + q0 + ln) * DH;
  v16h au0 = load_a16(Qub, g), au1 = load_a16(Qub + 32, g);
  v16h av0 = load_a16(Qvb, g), av1 = load_a16(Qvb + 32, g);

  float mrun[8], lrun[8];
  v8f oacc[4];
#pragma unroll
  for (int v = 0; v < 8; ++v) { mrun[v] = -1e30f; lrun[v] = 0.f; }
#pragma unroll
  for (int dt = 0; dt < 4; ++dt)
    oacc[dt] = (v8f){0.f, 0.f, 0.f, 0.f, 0.f, 0.f, 0.f, 0.f};

  const _Float16* Phh = Ph + (size_t)h * TN * DH;
  const int* mrow = maskp + ((size_t)b * TN + q0) * TN;

  for (int kt = 0; kt < TN; kt += 32) {
    v8f s0 = {0.f, 0.f, 0.f, 0.f, 0.f, 0.f, 0.f, 0.f};
    v8f s1 = {0.f, 0.f, 0.f, 0.f, 0.f, 0.f, 0.f, 0.f};
    const _Float16* Kb0 = Kh + ((size_t)bh * TN + kt + ln) * DH;
    const _Float16* Kb1 = Kb0 + 16 * DH;
    __builtin_prefetch(Kb0 + 32 * DH, 0, 3);
    s0 = wmma_f16(au0, load_b16(Kb0, g), s0);
    s0 = wmma_f16(au1, load_b16(Kb0 + 32, g), s0);
    s1 = wmma_f16(au0, load_b16(Kb1, g), s1);
    s1 = wmma_f16(au1, load_b16(Kb1 + 32, g), s1);

    int rbase = TN - 16 - q0 + kt;
#pragma unroll
    for (int j = 0; j < 3; ++j) {
      int pr = rbase + 16 * j + ln;
      pr = pr < 0 ? 0 : (pr > TN - 1 ? TN - 1 : pr);
      const _Float16* Pb = Phh + (size_t)pr * DH;
      v8f tj = {0.f, 0.f, 0.f, 0.f, 0.f, 0.f, 0.f, 0.f};
      tj = wmma_f16(av0, load_b16(Pb, g), tj);
      tj = wmma_f16(av1, load_b16(Pb + 32, g), tj);
#pragma unroll
      for (int v = 0; v < 8; ++v) bds[wid][v + 8 * g][16 * j + ln] = tj[v];
    }
    __syncthreads();

#pragma unroll
    for (int v = 0; v < 8; ++v) {
      int m = v + 8 * g;
      int qrow = q0 + m;
      int kA = kt + ln, kB = kt + 16 + ln;
      float bdA = (kA <= qrow) ? bds[wid][m][15 - m + ln] : 0.f;
      float bdB = (kB <= qrow) ? bds[wid][m][15 - m + 16 + ln] : 0.f;
      float sA = (s0[v] + bdA) * 0.125f;
      float sB = (s1[v] + bdB) * 0.125f;
      if (mrow[(size_t)m * TN + kA] == 0) sA = -1e30f;
      if (mrow[(size_t)m * TN + kB] == 0) sB = -1e30f;
      float mx = grp_max(fmaxf(sA, sB));
      float nm = fmaxf(mrun[v], mx);
      float pA = expf(sA - nm);
      float pB = expf(sB - nm);
      float rs = grp_sum(pA + pB);
      float sc = expf(mrun[v] - nm);
      lrun[v] = lrun[v] * sc + rs;
      mrun[v] = nm;
#pragma unroll
      for (int dt = 0; dt < 4; ++dt) oacc[dt][v] *= sc;
      ps[wid][m][ln] = (_Float16)pA;
      ps[wid][m][16 + ln] = (_Float16)pB;
    }
    __syncthreads();

    v16h ap = load_a16(&ps[wid][ln][0], g);
#pragma unroll
    for (int dt = 0; dt < 4; ++dt) {
      const _Float16* Vb = Vt + ((size_t)bh * DH + dt * 16 + ln) * TN + kt;
      oacc[dt] = wmma_f16(ap, load_b16(Vb, g), oacc[dt]);
    }
    __syncthreads();
  }

#pragma unroll
  for (int dt = 0; dt < 4; ++dt) {
#pragma unroll
    for (int v = 0; v < 8; ++v) {
      int m = v + 8 * g;
      float o = (lrun[v] > 0.f) ? (oacc[dt][v] / lrun[v]) : 0.f;
      Hbuf[((size_t)(b * TN) + q0 + m) * DM + h * DH + dt * 16 + ln] = o;
    }
  }
}


extern "C" void kernel_launch(void* const* d_in, const int* in_sizes, int n_in,
                              void* d_out, int out_size, void* d_ws, size_t ws_size,
                              hipStream_t stream) {
  const float* x = (const float*)d_in[0];
  const int* mask = (const int*)d_in[1];
  const float* pos = (const float*)d_in[2];
  const float* ln_g = (const float*)d_in[3];
  const float* ln_b = (const float*)d_in[4];
  const float* Wq = (const float*)d_in[5];
  const float* bq = (const float*)d_in[6];
  const float* Wk = (const float*)d_in[7];
  const float* bk = (const float*)d_in[8];
  const float* Wv = (const float*)d_in[9];
  const float* bv = (const float*)d_in[10];
  const float* Wp = (const float*)d_in[11];
  const float* bp = (const float*)d_in[12];
  const float* Wo = (const float*)d_in[13];
  const float* bo = (const float*)d_in[14];
  const float* pbu = (const float*)d_in[15];
  const float* pbv = (const float*)d_in[16];
  const int* bw = (const int*)d_in[17];
  float* out = (float*)d_out;

  const int M = BN * TN;
  uintptr_t p = (uintptr_t)d_ws;
  auto carve = [&](size_t bytes) -> void* {
    void* r = (void*)p;
    p += (bytes + 255) & ~(size_t)255;
    return r;
  };
  float* wsbuf = (float*)carve(5 * sizeof(float));
  int8_t* W8 = (int8_t*)carve((size_t)5 * DM * DM);
  int8_t* xq = (int8_t*)carve((size_t)M * DM);
  float* dqx = (float*)carve((size_t)M * sizeof(float));
  int8_t* pq = (int8_t*)carve((size_t)TN * DM);
  float* dqp = (float*)carve((size_t)TN * sizeof(float));
  float* Qf = (float*)carve((size_t)M * DM * sizeof(float));
  float* Kf = (float*)carve((size_t)M * DM * sizeof(float));
  float* Vf = (float*)carve((size_t)M * DM * sizeof(float));
  float* Pf = (float*)carve((size_t)TN * DM * sizeof(float));
  _Float16* Quh = (_Float16*)carve((size_t)M * DM * 2);
  _Float16* Qvh = (_Float16*)carve((size_t)M * DM * 2);
  _Float16* Kh = (_Float16*)carve((size_t)M * DM * 2);
  _Float16* Vt = (_Float16*)carve((size_t)M * DM * 2);
  _Float16* Ph = (_Float16*)carve((size_t)NH * TN * DH * 2);
  float* Hb = (float*)carve((size_t)M * DM * sizeof(float));
  int8_t* hq = (int8_t*)carve((size_t)M * DM);
  float* dqh = (float*)carve((size_t)M * sizeof(float));

  k_wscale<<<5, 256, 0, stream>>>(Wq, Wk, Wv, Wp, Wo, wsbuf);
  k_wquant<<<5 * DM * DM / 256, 256, 0, stream>>>(Wq, Wk, Wv, Wp, Wo, wsbuf, W8);

  k_lnq<<<M, 256, 0, stream>>>(x, ln_g, ln_b, bw, xq, dqx, 0);
  k_lnq<<<TN, 256, 0, stream>>>(pos, ln_g, ln_b, bw, pq, dqp, 1);

  int gemm_blocks_M = (M / 16) * (DM / 16) / 4;
  int gemm_blocks_T = (TN / 16) * (DM / 16) / 4;
  k_qgemm<<<gemm_blocks_M, 128, 0, stream>>>(xq, W8 + 0 * DM * DM, dqx, wsbuf, 0,
                                             bq, nullptr, nullptr, Qf, M);
  k_qgemm<<<gemm_blocks_M, 128, 0, stream>>>(xq, W8 + 1 * DM * DM, dqx, wsbuf, 1,
                                             bk, nullptr, nullptr, Kf, M);
  k_qgemm<<<gemm_blocks_M, 128, 0, stream>>>(xq, W8 + 2 * DM * DM, dqx, wsbuf, 2,
                                             bv, nullptr, nullptr, Vf, M);
  k_qgemm<<<gemm_blocks_T, 128, 0, stream>>>(pq, W8 + 3 * DM * DM, dqp, wsbuf, 3,
                                             bp, nullptr, nullptr, Pf, TN);

  k_pack<<<M * DM / 256, 256, 0, stream>>>(Qf, Kf, Vf, Pf, pbu, pbv, Quh, Qvh, Kh,
                                           Vt, Ph);

  k_attn<<<BN * NH * (TN / 16) / 4, 128, 0, stream>>>(Quh, Qvh, Kh, Vt, Ph, mask,
                                                      Hb);

  k_lnq<<<M, 256, 0, stream>>>(Hb, ln_g, ln_b, bw, hq, dqh, 1);
  k_qgemm<<<gemm_blocks_M, 128, 0, stream>>>(hq, W8 + 4 * DM * DM, dqh, wsbuf, 4,
                                             bo, x, mask, out, M);
}
